// AnisoMultiGaussSpatialConv_66322884985176
// MI455X (gfx1250) — matrix operand order repaired and checked by reading, not compiled, not run on hardware
//
#include <hip/hip_runtime.h>
#include <hip/hip_bf16.h>

// ---------------------------------------------------------------------------
// AnisoMultiGaussSpatialConv for MI455X (gfx1250, wave32, WMMA + TDM)
//
// dist2[n,m] = a_n . b_m  with
//   a_n = [x x^T (9), x (3), 1, 0-pad...]                     (32 f16)
//   b_m = [Gamma (9), -2*Gamma y (3), y^T Gamma y, 0-pad...]  (32 f16)
// K[n,m] = 0.2 e^{-200 d} + 0.3 e^{-50 d} + 0.5 e^{-12.5 d}
//        = 0.2 t^16 + 0.3 t^4 + 0.5 t,   t = e^{-12.5 d}   (ONE v_exp_f32)
// out    = K @ y_fea
// Both GEMMs run on v_wmma_f32_16x16x32_f16; b-feature tiles staged into LDS
// by the Tensor Data Mover (6-arg builtin on this toolchain).
// ---------------------------------------------------------------------------

typedef __attribute__((ext_vector_type(16))) _Float16     v16h;
typedef __attribute__((ext_vector_type(8)))  float        v8f;
typedef __attribute__((ext_vector_type(8)))  unsigned int v8u;
typedef __attribute__((ext_vector_type(2)))  _Float16     h2;
typedef __attribute__((ext_vector_type(4)))  unsigned int u32x4;
typedef __attribute__((ext_vector_type(8)))  int          i32x8;
typedef __attribute__((ext_vector_type(4)))  int          i32x4;

#if defined(__has_builtin)
#if __has_builtin(__builtin_amdgcn_tensor_load_to_lds) && \
    __has_builtin(__builtin_amdgcn_s_wait_tensorcnt)
#define USE_TDM 1
#endif
#endif

#define B_    2
#define N_    4096
#define M_    4096
#define F_    64
#define KF    32      // padded feature length (13 used)
#define KFP   (KF/2)  // packed f16 pairs per feature vector
#define NTILE 128     // n rows per block (8 waves x 16)
#define MTILE 32      // m columns per iteration (WMMA K for GEMM2)

// t = exp(-d/(2*0.2^2)) = exp2(-d * log2(e)*12.5)
#define C_S020 18.033688011112043f

// ---------------------------------------------------------------------------
// Kernel 1: build padded f16 feature vectors for both sides.
// ---------------------------------------------------------------------------
__global__ __launch_bounds__(256) void aniso_feat_kernel(
    const float* __restrict__ x, const float* __restrict__ y,
    const float* __restrict__ gamma,
    _Float16* __restrict__ aF, _Float16* __restrict__ bF)
{
    int tid = blockIdx.x * blockDim.x + threadIdx.x;
    if (tid >= B_ * N_) return;
    int b = tid / N_;
    int i = tid - b * N_;

    // ---- a-side (query points x) ----
    float xv[3];
    #pragma unroll
    for (int d = 0; d < 3; ++d) xv[d] = x[(b * N_ + i) * 3 + d];
    _Float16 a[KF];
    #pragma unroll
    for (int k = 0; k < KF; ++k) a[k] = (_Float16)0.0f;
    #pragma unroll
    for (int d = 0; d < 3; ++d)
        #pragma unroll
        for (int e = 0; e < 3; ++e) a[d * 3 + e] = (_Float16)(xv[d] * xv[e]);
    #pragma unroll
    for (int d = 0; d < 3; ++d) a[9 + d] = (_Float16)xv[d];
    a[12] = (_Float16)1.0f;
    #pragma unroll
    for (int k = 0; k < KF; ++k) aF[(b * N_ + i) * KF + k] = a[k];

    // ---- b-side (source points y with metric Gamma) ----
    float yv[3];
    #pragma unroll
    for (int d = 0; d < 3; ++d) yv[d] = y[(b * M_ + i) * 3 + d];
    float G[9];
    #pragma unroll
    for (int p = 0; p < 9; ++p) G[p] = gamma[(b * M_ + i) * 9 + p];
    float Gy[3];
    #pragma unroll
    for (int d = 0; d < 3; ++d)
        Gy[d] = G[d * 3 + 0] * yv[0] + G[d * 3 + 1] * yv[1] + G[d * 3 + 2] * yv[2];
    float yGy = Gy[0] * yv[0] + Gy[1] * yv[1] + Gy[2] * yv[2];

    _Float16 bb[KF];
    #pragma unroll
    for (int k = 0; k < KF; ++k) bb[k] = (_Float16)0.0f;
    #pragma unroll
    for (int p = 0; p < 9; ++p) bb[p] = (_Float16)G[p];
    #pragma unroll
    for (int d = 0; d < 3; ++d) bb[9 + d] = (_Float16)(-2.0f * Gy[d]);
    bb[12] = (_Float16)yGy;
    #pragma unroll
    for (int k = 0; k < KF; ++k) bF[(b * M_ + i) * KF + k] = bb[k];
}

// ---------------------------------------------------------------------------
// Kernel 2: main fused GEMM -> exp-mix -> GEMM.
// Block = 256 threads = 8 waves; wave w owns n rows [n0b + 16w, +16), all 64 f.
// ---------------------------------------------------------------------------
__global__ __launch_bounds__(256) void aniso_conv_kernel(
    const _Float16* __restrict__ aF, const _Float16* __restrict__ bF,
    const float* __restrict__ y_fea, float* __restrict__ out)
{
    __shared__ unsigned int ldsB[MTILE * KFP];        // 32 m x 16 packed pairs (2 KB)
    __shared__ unsigned int ldsY[F_ * (MTILE / 2)];   // 64 f x 16 packed m-pairs (4 KB)
    __shared__ _Float16     ldsK[8 * 16 * MTILE];     // per-wave 16n x 32m K tile (8 KB)

    const int b    = blockIdx.y;
    const int n0b  = blockIdx.x * NTILE;
    const int lane = threadIdx.x & 31;
    const int wave = threadIdx.x >> 5;
    const int hi   = (lane >> 4) & 1;   // lane half (WMMA layout split)
    const int lo   = lane & 15;

    const unsigned int* aFu = (const unsigned int*)aF;
    const unsigned int* bFu = (const unsigned int*)bF;

    // Hoisted GEMM1 A operand: 16x32 f16, contiguous k order (same as B operand)
    //  lane(0-15)=row n ; VGPR v holds K pair  (v + hi*8)
    const int nrow = n0b + wave * 16 + lo;
    v8u au;
    #pragma unroll
    for (int v = 0; v < 8; ++v) {
        int pair = v + hi * 8;
        au[v] = aFu[(b * N_ + nrow) * KFP + pair];
    }
    v16h aFeat = __builtin_bit_cast(v16h, au);

    v8f zero = {};
    v8f acc[4] = {zero, zero, zero, zero};

    for (int mt = 0; mt < M_; mt += MTILE) {
#ifdef USE_TDM
        // ---- TDM: DMA the 2 KB b-feature tile into LDS (wave 0 issues) ----
        if (wave == 0) {
            unsigned long long ga =
                (unsigned long long)(const void*)&bFu[(b * M_ + mt) * KFP];
            unsigned ldsOff = (unsigned)(unsigned long long)(void*)ldsB;
            u32x4 g0;
            g0[0] = 1u;                                   // count=1, user mode
            g0[1] = ldsOff;                               // lds_addr
            g0[2] = (unsigned)(ga & 0xffffffffu);         // global_addr[31:0]
            g0[3] = (unsigned)((ga >> 32) & 0x01ffffffu)  // global_addr[56:32]
                  | 0x80000000u;                          // D# type = 2 (image)
            i32x8 g1 = {};
            g1[0] = (int)(2u << 16);     // data_size=2 (4B), workgroup_mask=0
            g1[1] = (int)(512u << 16);   // tensor_dim0[15:0] = 512 dwords
            g1[2] = 0;                   // tensor_dim0[31:16]=0, tensor_dim1=0
            g1[3] = (int)(512u << 16);   // tile_dim0 = 512 dwords (1-D tile)
            g1[4] = 0;                   // tile_dim1=0 (unused), tile_dim2=0
            g1[5] = (int)512;            // tensor_dim0_stride[31:0]
            g1[6] = 0;
            g1[7] = 0;
            i32x4 gz4 = {};
            i32x8 gz8 = {};
            __builtin_amdgcn_tensor_load_to_lds(g0, g1, gz4, gz4, gz8, 0);
        }
#else
        for (int u = threadIdx.x; u < MTILE * KFP; u += 256) {
            int m_loc = u >> 4;
            int j     = u & 15;
            ldsB[u] = bFu[(b * M_ + mt + m_loc) * KFP + j];
        }
#endif
        // ---- cooperative stage: y_fea, f32 -> f16, transposed [f][m-pairs] ----
        for (int u = threadIdx.x; u < F_ * (MTILE / 2); u += 256) {
            int f  = u >> 4;
            int jp = u & 15;
            int m0 = mt + jp * 2;
            h2 p;
            p.x = (_Float16)y_fea[(b * M_ + m0) * F_ + f];
            p.y = (_Float16)y_fea[(b * M_ + m0 + 1) * F_ + f];
            ldsY[u] = __builtin_bit_cast(unsigned int, p);
        }
        if (mt + MTILE < M_) {
            __builtin_prefetch(&bFu[(b * M_ + mt + MTILE) * KFP], 0, 1);
            __builtin_prefetch(&y_fea[(b * M_ + mt + MTILE) * F_], 0, 1);
        }
#ifdef USE_TDM
        if (wave == 0) __builtin_amdgcn_s_wait_tensorcnt(0);
#endif
        __syncthreads();

        // ---- GEMM1: dist2 tiles (16 n x 32 m), one WMMA per 16x16 subtile ----
        #pragma unroll
        for (int t = 0; t < 2; ++t) {
            // B operand: 32x16 f16, contiguous k order (same as A): lane=col m, VGPR v = K pair (v + hi*8)
            v8u bu;
            #pragma unroll
            for (int v = 0; v < 8; ++v)
                bu[v] = ldsB[(t * 16 + lo) * KFP + v + hi * 8];
            v16h bFeat = __builtin_bit_cast(v16h, bu);

            v8f dz = {};
            dz = __builtin_amdgcn_wmma_f32_16x16x32_f16(
                false, aFeat, false, bFeat, (short)0, dz, false, false);

            // ---- exp mixture epilogue: sigma set is {s,2s,4s} so all three
            //      Gaussians are powers of one exp:  K = .5 t + .3 t^4 + .2 t^16
            #pragma unroll
            for (int r = 0; r < 8; ++r) {
                float d   = dz[r];
                float t1  = __builtin_amdgcn_exp2f(-d * C_S020); // exp(-12.5 d)
                float t2  = t1 * t1;
                float t4  = t2 * t2;                             // exp(-50 d)
                float t8  = t4 * t4;
                float t16 = t8 * t8;                             // exp(-200 d)
                float kv  = 0.5f * t1 + 0.3f * t4 + 0.2f * t16;
                int n_loc = r + hi * 8;
                int m_loc = t * 16 + lo;
                ldsK[(wave * 16 + n_loc) * MTILE + m_loc] = (_Float16)kv;
            }
        }
        // ldsK is wave-private; same-wave DS ops are processed in order by the
        // LDS (DScnt in-order), so no block barrier is needed — only stop the
        // compiler from reordering the loads above the stores.
        asm volatile("" ::: "memory");

        // ---- GEMM2: acc += K(16n x 32m) @ y_fea(32m x 64f) ----
        const unsigned int* ldsKu = (const unsigned int*)ldsK;
        v8u ku;
        #pragma unroll
        for (int v = 0; v < 8; ++v) {
            int pair = v + hi * 8;   // contiguous k pairs, same order as yB
            ku[v] = ldsKu[(wave * 16 + lo) * (MTILE / 2) + pair];
        }
        v16h kA = __builtin_bit_cast(v16h, ku);

        #pragma unroll
        for (int t = 0; t < 4; ++t) {
            v8u yu;
            #pragma unroll
            for (int v = 0; v < 8; ++v)
                yu[v] = ldsY[(t * 16 + lo) * (MTILE / 2) + v + hi * 8];
            v16h yB = __builtin_bit_cast(v16h, yu);
            acc[t] = __builtin_amdgcn_wmma_f32_16x16x32_f16(
                false, kA, false, yB, (short)0, acc[t], false, false);
        }
        __syncthreads();
    }

    // ---- write out: D layout -> out[b][n][f] ----
    #pragma unroll
    for (int t = 0; t < 4; ++t) {
        #pragma unroll
        for (int r = 0; r < 8; ++r) {
            int n = n0b + wave * 16 + r + hi * 8;
            int f = t * 16 + lo;
            out[(b * N_ + n) * F_ + f] = acc[t][r];
        }
    }
}

// ---------------------------------------------------------------------------
extern "C" void kernel_launch(void* const* d_in, const int* in_sizes, int n_in,
                              void* d_out, int out_size, void* d_ws, size_t ws_size,
                              hipStream_t stream) {
    const float* x     = (const float*)d_in[0];
    const float* y     = (const float*)d_in[1];
    const float* y_fea = (const float*)d_in[2];
    const float* gamma = (const float*)d_in[3];
    float* out = (float*)d_out;

    _Float16* aF = (_Float16*)d_ws;
    _Float16* bF = (_Float16*)((char*)d_ws + (size_t)B_ * N_ * KF * sizeof(_Float16));

    dim3 g1((B_ * N_ + 255) / 256);
    aniso_feat_kernel<<<g1, 256, 0, stream>>>(x, y, gamma, aF, bF);

    dim3 g2(N_ / NTILE, B_);
    aniso_conv_kernel<<<g2, 256, 0, stream>>>(aF, bF, y_fea, out);
}
